// GCN_12635793785487
// MI455X (gfx1250) — matrix operand order repaired and checked by reading, not compiled, not run on hardware
//
#include <hip/hip_runtime.h>
#include <hip/hip_bf16.h>

#define N_NODES  50000
#define N_EDGES  800000
#define N_GRAPHS 512
#define D_H      64
#define D_OUT    32
#define LN_EPS   1e-5f

typedef __attribute__((ext_vector_type(2)))  float    v2f;
typedef __attribute__((ext_vector_type(8)))  float    v8f;
typedef __attribute__((ext_vector_type(16))) _Float16 v16h;

#if defined(__has_builtin)
#if __has_builtin(__builtin_amdgcn_wmma_f32_16x16x4_f32)
#define USE_F32_WMMA 1
#endif
#endif

// ---------------------------------------------------------------------------
// GEMM: C[M,NCOLS] = A[M,64] @ B[64,NCOLS] (+ bias). One wave per 16x16 tile.
// Block = NCOLS/16 waves; grid.x = M/16 (M is an exact multiple of 16).
// No divergence inside the kernel => EXEC all-ones as WMMA requires.
//
// f32 WMMA 16x16x4 fragment layout (ISA 7.12.2):
//   A (16x4):  lane m=lane&15, khalf=lane>>4; VGPR j holds A[m][k0+2*khalf+j]
//   B (4x16):  lane n=lane&15; VGPR j holds B[k0+2*khalf+j][n] (mirrors C/D split)
//   C/D(16x16): VGPR i holds D[i + 8*(lane>>4)][lane&15]
// ---------------------------------------------------------------------------
template <int NCOLS, bool BIAS>
__global__ __launch_bounds__(32 * (NCOLS / 16)) void gcn_gemm_wmma(
    const float* __restrict__ A, const float* __restrict__ B,
    const float* __restrict__ bias, float* __restrict__ C) {
  const int lane = threadIdx.x & 31;
  const int wave = threadIdx.x >> 5;
  const int m    = lane & 15;
  const int half = lane >> 4;
  const int c0   = wave * 16;
  const int r0   = blockIdx.x * 16;
  const int row  = r0 + m;

  v8f acc = {0.f, 0.f, 0.f, 0.f, 0.f, 0.f, 0.f, 0.f};

#if USE_F32_WMMA
  #pragma unroll
  for (int k0 = 0; k0 < 64; k0 += 4) {
    const int ka = k0 + 2 * half;
    v2f a, b;
    a.x = A[row * 64 + ka];
    a.y = A[row * 64 + ka + 1];
    b.x = B[ka * NCOLS + c0 + m];
    b.y = B[(ka + 1) * NCOLS + c0 + m];
    acc = __builtin_amdgcn_wmma_f32_16x16x4_f32(false, a, false, b,
                                                (short)0, acc, false, false);
  }
#else
  // Fallback: codegen-confirmed f16 WMMA with on-the-fly conversion.
  for (int kstep = 0; kstep < 2; ++kstep) {
    v16h af, bf;
    #pragma unroll
    for (int e = 0; e < 16; ++e) {     // 16-bit A 16x32 layout (ISA 7.12.2)
      int v = e >> 1, j = e & 1, grp = v >> 2;
      int K = 32 * kstep + 16 * grp + 8 * half + ((v & 3) * 2 + j);
      af[e] = (_Float16)A[row * 64 + K];
    }
    #pragma unroll
    for (int e = 0; e < 16; ++e) {     // B 32x16: same K layout as A (16*grp + 8*half + 2*(v&3) + j)
      int v = e >> 1, j = e & 1, grp = v >> 2;
      int K = 32 * kstep + 16 * grp + 8 * half + ((v & 3) * 2 + j);
      bf[e] = (_Float16)B[K * NCOLS + c0 + m];
    }
    acc = __builtin_amdgcn_wmma_f32_16x16x32_f16(false, af, false, bf,
                                                 (short)0, acc, false, false);
  }
#endif

  #pragma unroll
  for (int i = 0; i < 8; ++i) {
    float v = acc[i];
    if (BIAS) v += bias[c0 + m];
    C[(r0 + i + 8 * half) * NCOLS + c0 + m] = v;
  }
}

// ---------------------------------------------------------------------------
// GCN normalization precompute
// ---------------------------------------------------------------------------
__global__ void k_deg_init(float* __restrict__ deg) {
  int i = blockIdx.x * blockDim.x + threadIdx.x;
  if (i < N_NODES) deg[i] = 1.0f;  // self-loop
}

__global__ void k_deg_accum(const int* __restrict__ dst, float* __restrict__ deg) {
  int i = blockIdx.x * blockDim.x + threadIdx.x;
  if (i < N_EDGES) atomicAdd(&deg[dst[i]], 1.0f);
}

__global__ void k_dinv(float* __restrict__ deg) {  // in place: deg -> deg^-1/2
  int i = blockIdx.x * blockDim.x + threadIdx.x;
  if (i < N_NODES) deg[i] = rsqrtf(deg[i]);        // deg >= 1 always
}

__global__ void k_enorm(const int* __restrict__ src, const int* __restrict__ dst,
                        const float* __restrict__ dinv, float* __restrict__ enorm) {
  int i = blockIdx.x * blockDim.x + threadIdx.x;
  if (i < N_EDGES) enorm[i] = dinv[src[i]] * dinv[dst[i]];
}

// agg[n,f] = dinv[n]^2 * hW[n,f]   (self-loop message; also zero-inits agg)
__global__ void k_selfloop(const float* __restrict__ hW, const float* __restrict__ dinv,
                           float* __restrict__ agg) {
  int idx = blockIdx.x * blockDim.x + threadIdx.x;
  if (idx >= N_NODES * 64) return;
  float di = dinv[idx >> 6];
  agg[idx] = di * di * hW[idx];
}

// agg[dst,f] += enorm[e] * hW[src,f]  (L2-resident f32 atomics; data < 192MB L2)
__global__ void k_edge_scatter(const float* __restrict__ hW, const int* __restrict__ src,
                               const int* __restrict__ dst, const float* __restrict__ enorm,
                               float* __restrict__ agg) {
  int idx = blockIdx.x * blockDim.x + threadIdx.x;
  if (idx >= N_EDGES * 64) return;
  int e = idx >> 6, f = idx & 63;
  int s = src[e], d = dst[e];
  atomicAdd(&agg[d * 64 + f], enorm[e] * hW[s * 64 + f]);
}

// h_out = relu(layernorm(agg + bias) * w + b)   — one node per wave32
__global__ void k_bias_ln_relu(const float* __restrict__ agg, const float* __restrict__ bias,
                               const float* __restrict__ lnw, const float* __restrict__ lnb,
                               float* __restrict__ out) {
  int node = (blockIdx.x * blockDim.x + threadIdx.x) >> 5;
  int lane = threadIdx.x & 31;
  if (node >= N_NODES) return;  // uniform per wave
  const float* rowp = agg + (size_t)node * 64;
  float v0 = rowp[lane] + bias[lane];
  float v1 = rowp[lane + 32] + bias[lane + 32];
  float s = v0 + v1;
  #pragma unroll
  for (int off = 16; off > 0; off >>= 1) s += __shfl_xor(s, off, 32);
  float mu = s * (1.0f / 64.0f);
  float d0 = v0 - mu, d1 = v1 - mu;
  float q = d0 * d0 + d1 * d1;
  #pragma unroll
  for (int off = 16; off > 0; off >>= 1) q += __shfl_xor(q, off, 32);
  float rstd = rsqrtf(q * (1.0f / 64.0f) + LN_EPS);
  float* orow = out + (size_t)node * 64;
  orow[lane]      = fmaxf(0.f, d0 * rstd * lnw[lane] + lnb[lane]);
  orow[lane + 32] = fmaxf(0.f, d1 * rstd * lnw[lane + 32] + lnb[lane + 32]);
}

// ---------------------------------------------------------------------------
// Global mean pool
// ---------------------------------------------------------------------------
__global__ void k_zero_pool(float* __restrict__ pool, float* __restrict__ cnt) {
  int i = blockIdx.x * blockDim.x + threadIdx.x;
  if (i < N_GRAPHS * 64) pool[i] = 0.f;
  if (i < N_GRAPHS) cnt[i] = 0.f;
}

__global__ void k_pool_accum(const float* __restrict__ h, const int* __restrict__ batch,
                             float* __restrict__ pool, float* __restrict__ cnt) {
  int idx = blockIdx.x * blockDim.x + threadIdx.x;
  if (idx >= N_NODES * 64) return;
  int n = idx >> 6, f = idx & 63;
  int g = batch[n];
  atomicAdd(&pool[g * 64 + f], h[idx]);
  if (f == 0) atomicAdd(&cnt[g], 1.0f);
}

__global__ void k_pool_div(float* __restrict__ pool, const float* __restrict__ cnt) {
  int i = blockIdx.x * blockDim.x + threadIdx.x;
  if (i < N_GRAPHS * 64) pool[i] /= fmaxf(cnt[i >> 6], 1.0f);
}

// ---------------------------------------------------------------------------
// Launch
// ---------------------------------------------------------------------------
static inline int cdiv(int a, int b) { return (a + b - 1) / b; }

static void run_layer(const float* hin, const float* W, const float* b,
                      const float* lnw, const float* lnb,
                      const float* dinv, const int* src, const int* dst,
                      const float* enorm, float* hW, float* agg, float* hout,
                      hipStream_t stream) {
  gcn_gemm_wmma<64, false><<<N_NODES / 16, 128, 0, stream>>>(hin, W, nullptr, hW);
  k_selfloop<<<cdiv(N_NODES * 64, 256), 256, 0, stream>>>(hW, dinv, agg);
  k_edge_scatter<<<cdiv(N_EDGES * 64, 256), 256, 0, stream>>>(hW, src, dst, enorm, agg);
  k_bias_ln_relu<<<cdiv(N_NODES * 32, 256), 256, 0, stream>>>(agg, b, lnw, lnb, hout);
}

extern "C" void kernel_launch(void* const* d_in, const int* in_sizes, int n_in,
                              void* d_out, int out_size, void* d_ws, size_t ws_size,
                              hipStream_t stream) {
  (void)in_sizes; (void)n_in; (void)out_size; (void)ws_size;

  const float* x       = (const float*)d_in[0];
  const int*   eidx    = (const int*)d_in[1];
  const int*   batch   = (const int*)d_in[2];
  const float* W_in    = (const float*)d_in[3];
  const float* b_in    = (const float*)d_in[4];
  const float* ln_in_w = (const float*)d_in[5];
  const float* ln_in_b = (const float*)d_in[6];
  const float* W_h     = (const float*)d_in[7];   // [2,64,64]
  const float* b_h     = (const float*)d_in[8];   // [2,64]
  const float* ln_h_w  = (const float*)d_in[9];   // [2,64]
  const float* ln_h_b  = (const float*)d_in[10];  // [2,64]
  const float* W_out   = (const float*)d_in[11];  // [64,32]
  const float* b_out   = (const float*)d_in[12];  // [32]

  const int* src = eidx;            // edge_index[0] = message sources
  const int* dst = eidx + N_EDGES;  // edge_index[1] = aggregation targets

  float* ws    = (float*)d_ws;
  float* dinv  = ws;                       // N        (deg, then deg^-1/2 in place)
  float* enorm = dinv + N_NODES;           // E
  float* hW    = enorm + N_EDGES;          // N*64
  float* agg   = hW + (size_t)N_NODES*64;  // N*64
  float* hbuf  = agg + (size_t)N_NODES*64; // N*64
  float* pool  = hbuf + (size_t)N_NODES*64;// G*64
  float* cnt   = pool + N_GRAPHS * 64;     // G

  // gcn_norm
  k_deg_init<<<cdiv(N_NODES, 256), 256, 0, stream>>>(dinv);
  k_deg_accum<<<cdiv(N_EDGES, 256), 256, 0, stream>>>(dst, dinv);
  k_dinv<<<cdiv(N_NODES, 256), 256, 0, stream>>>(dinv);
  k_enorm<<<cdiv(N_EDGES, 256), 256, 0, stream>>>(src, dst, dinv, enorm);

  // 3 conv layers (in + 2 hidden)
  run_layer(x,    W_in,       b_in,       ln_in_w,      ln_in_b,      dinv, src, dst, enorm, hW, agg, hbuf, stream);
  run_layer(hbuf, W_h,        b_h,        ln_h_w,       ln_h_b,       dinv, src, dst, enorm, hW, agg, hbuf, stream);
  run_layer(hbuf, W_h + 64*64, b_h + 64,  ln_h_w + 64,  ln_h_b + 64,  dinv, src, dst, enorm, hW, agg, hbuf, stream);

  // global mean pool + output projection (WMMA head, fused bias)
  k_zero_pool<<<cdiv(N_GRAPHS * 64, 256), 256, 0, stream>>>(pool, cnt);
  k_pool_accum<<<cdiv(N_NODES * 64, 256), 256, 0, stream>>>(hbuf, batch, pool, cnt);
  k_pool_div<<<cdiv(N_GRAPHS * 64, 256), 256, 0, stream>>>(pool, cnt);
  gcn_gemm_wmma<D_OUT, true><<<N_GRAPHS / 16, 64, 0, stream>>>(pool, W_out, b_out, (float*)d_out);
}
